// MESRNN_15556371546759
// MI455X (gfx1250) — matrix operand order repaired and checked by reading, not compiled, not run on hardware
//
#include <hip/hip_runtime.h>

typedef __attribute__((ext_vector_type(16))) _Float16 v16h;
typedef __attribute__((ext_vector_type(8)))  _Float16 v8h;
typedef __attribute__((ext_vector_type(8)))  float    v8f;
typedef unsigned int u32x4 __attribute__((ext_vector_type(4)));
typedef int          i32x8 __attribute__((ext_vector_type(8)));
typedef int          i32x4 __attribute__((ext_vector_type(4)));

#define NN_E 6
#define ROWS 16
#define WAVES_PER_BLOCK 8
#define BLOCK_THREADS 256
#define N_STAGES 10                             // 6 edge + 4 node weight stages

// ---- workspace (f16 halves): pre-packed B fragments, stage s at ws + s*32768 ----
// edge stage e: [ct16][src2][kk2][lane32][j16] ; node stage hct: [g4][kk16][lane32][j16]
#define STAGE_HALVES   32768                    // 64KB per stage
#define WS_NODE_OFF    (NN_E * STAGE_HALVES)    // 196608
#define WS_TOTAL_HALVES (N_STAGES * STAGE_HALVES)

// ---- LDS layout (bytes): double-buffered weight window + per-wave scratch ----
#define LDS_W_BYTES    65536
#define CONCAT_STRIDE  520                      // 512 cols + 8 pad (halves)
#define CONCAT_HALVES_PER_WAVE (ROWS * CONCAT_STRIDE)
#define EMB_STRIDE     72
#define DEC_STRIDE     68
#define LDS_CONCAT_OFF (2 * LDS_W_BYTES)        // 131072
#define LDS_EMB_OFF    (LDS_CONCAT_OFF + WAVES_PER_BLOCK * CONCAT_HALVES_PER_WAVE * 2)
#define LDS_DEC_OFF    (LDS_EMB_OFF + WAVES_PER_BLOCK * ROWS * EMB_STRIDE * 2)
#define LDS_TOTAL      (LDS_DEC_OFF + WAVES_PER_BLOCK * ROWS * DEC_STRIDE * 4)  // 317440 <= 320KB

// ============ weight packing kernel: f32 -> f16 B-fragment images ============
__global__ void mesrnn_pack_weights(const float* __restrict__ e_wih,
                                    const float* __restrict__ e_whh,
                                    const float* __restrict__ n_wih,
                                    const float* __restrict__ n_whh,
                                    _Float16* __restrict__ ws) {
  int tid = blockIdx.x * blockDim.x + threadIdx.x;
  if (tid >= WS_TOTAL_HALVES) return;
  float v;
  if (tid < WS_NODE_OFF) {
    int e    = tid >> 15;            // /32768
    int r    = tid & 32767;
    int j    = r & 15;
    int lane = (r >> 4) & 31;
    int kk   = (r >> 9) & 1;
    int src  = (r >> 10) & 1;
    int ct   = r >> 11;              // 0..15
    int nn   = ct * 16 + (lane & 15);
    int k    = kk * 32 + ((lane >> 4) << 3) + (j & 7) + ((j >> 3) << 4);
    const float* Wm = src ? e_whh : e_wih;
    v = Wm[((size_t)e * 256 + nn) * 64 + k];
  } else {
    int t    = tid - WS_NODE_OFF;
    int j    = t & 15;
    int lane = (t >> 4) & 31;
    int kk   = (t >> 9) & 15;
    int g    = (t >> 13) & 3;
    int hct  = t >> 15;              // 0..3
    int ct   = g * 4 + hct;
    int nn   = ct * 16 + (lane & 15);
    int k    = kk * 32 + ((lane >> 4) << 3) + (j & 7) + ((j >> 3) << 4);
    v = (k < 448) ? n_wih[(size_t)nn * 448 + k]
                  : n_whh[(size_t)nn * 64 + (k - 448)];
  }
  ws[tid] = (_Float16)v;
}

// ============ helpers ============
__device__ __forceinline__ float fsig(float x) { return 1.0f / (1.0f + __expf(-x)); }
__device__ __forceinline__ float ftanh_(float x) { return 1.0f - 2.0f / (1.0f + __expf(2.0f * x)); }

__device__ __forceinline__ v16h combine16(v8h lo, v8h hi) {
  v16h r;
#pragma unroll
  for (int j = 0; j < 8; ++j) { r[j] = lo[j]; r[8 + j] = hi[j]; }
  return r;
}

// A fragment (16x32 f16): lane 0-15 -> row m, K {0..7,16..23}; lane16-31 -> K {8..15,24..31}
__device__ __forceinline__ v16h load_afrag_lds(const _Float16* rowkp) {
  v8h lo = *(const v8h*)(rowkp);
  v8h hi = *(const v8h*)(rowkp + 16);
  return combine16(lo, hi);
}

__device__ __forceinline__ v16h load_afrag_gf32(const float* __restrict__ row, int kk, int kbase) {
  const float* p = row + kk * 32 + kbase;
  float4 a0 = ((const float4*)p)[0];
  float4 a1 = ((const float4*)p)[1];
  float4 b0 = ((const float4*)(p + 16))[0];
  float4 b1 = ((const float4*)(p + 16))[1];
  v16h r;
  r[0] = (_Float16)a0.x; r[1] = (_Float16)a0.y; r[2] = (_Float16)a0.z; r[3] = (_Float16)a0.w;
  r[4] = (_Float16)a1.x; r[5] = (_Float16)a1.y; r[6] = (_Float16)a1.z; r[7] = (_Float16)a1.w;
  r[8]  = (_Float16)b0.x; r[9]  = (_Float16)b0.y; r[10] = (_Float16)b0.z; r[11] = (_Float16)b0.w;
  r[12] = (_Float16)b1.x; r[13] = (_Float16)b1.y; r[14] = (_Float16)b1.z; r[15] = (_Float16)b1.w;
  return r;
}

// B fragment: lane's 16 halves contiguous at [(frag*32+lane)*16]
__device__ __forceinline__ v16h load_bfrag(const _Float16* W, int frag, int lane) {
  const _Float16* p = W + (((size_t)frag * 32 + lane) << 4);
  v8h lo = *(const v8h*)p;
  v8h hi = *(const v8h*)(p + 8);
  return combine16(lo, hi);
}

__device__ __forceinline__ v8f splat8(float x) {
  v8f a;
#pragma unroll
  for (int j = 0; j < 8; ++j) a[j] = x;
  return a;
}

// ---- Tensor Data Mover: DMA one 64KB weight stage (global -> LDS) ----
// D# group0: count=1 | lds_addr | 57b global addr | type=2
// D# group1: data_size=8B, tile 8192x1, tensor_dim0=8192, stride0=8192
__device__ __forceinline__ void tdm_load_stage(const _Float16* src, unsigned lds_byte_off) {
  unsigned long long ga = (unsigned long long)(uintptr_t)src;
  u32x4 g0;
  g0[0] = 1u;                                           // count=1 (valid descriptor)
  g0[1] = lds_byte_off;                                 // lds_addr [63:32]
  g0[2] = (unsigned)(ga & 0xFFFFFFFFu);                 // global_addr [95:64]
  g0[3] = (unsigned)((ga >> 32) & 0x01FFFFFFu) | (2u << 30);  // addr[120:96] | type=2
  i32x8 g1;
  g1[0] = (3 << 16);                                    // data_size=3 (8 bytes)
  g1[1] = (8192 & 0xFFFF) << 16;                        // tensor_dim0 lo16 (bits 63:48)
  g1[2] = ((8192 >> 16) & 0xFFFF) | (1 << 16);          // tensor_dim0 hi | tensor_dim1 lo=1
  g1[3] = (8192 << 16);                                 // tensor_dim1 hi=0 | tile_dim0=8192
  g1[4] = 1;                                            // tile_dim1=1, tile_dim2=0
  g1[5] = 8192;                                         // tensor_dim0_stride lo32
  g1[6] = 0;                                            // stride0 hi | stride1 lo
  g1[7] = 0;                                            // stride1 hi
  i32x4 g2 = {0, 0, 0, 0};
  i32x4 g3 = {0, 0, 0, 0};
  i32x8 g4 = {0, 0, 0, 0, 0, 0, 0, 0};
  __builtin_amdgcn_tensor_load_to_lds(g0, g1, g2, g3, g4, 0);
}

#define WMMA(A, B, C) __builtin_amdgcn_wmma_f32_16x16x32_f16(false, (A), false, (B), (short)0, (C), false, false)

// ============ main kernel ============
__global__ void __launch_bounds__(BLOCK_THREADS, 1)
mesrnn_main(const float* __restrict__ node_pos, const float* __restrict__ edge_inputs,
            const float* __restrict__ edge_h,   const float* __restrict__ edge_c,
            const float* __restrict__ node_h,   const float* __restrict__ node_c,
            const float* __restrict__ e_emb_w,  const float* __restrict__ e_emb_b,
            const float* __restrict__ e_bih,    const float* __restrict__ e_bhh,
            const float* __restrict__ n_emb_w,  const float* __restrict__ n_emb_b,
            const float* __restrict__ n_bih,    const float* __restrict__ n_bhh,
            const float* __restrict__ dec_w,    const float* __restrict__ dec_b,
            const _Float16* __restrict__ ws,
            float* __restrict__ out_pos, float* __restrict__ out_eh,
            float* __restrict__ out_ec,  float* __restrict__ out_nh,
            float* __restrict__ out_nc,  int N) {
  extern __shared__ char smem[];
  const int tid  = threadIdx.x;
  const int wave = tid >> 5;
  const int lane = tid & 31;
  _Float16* concat = (_Float16*)(smem + LDS_CONCAT_OFF) + wave * CONCAT_HALVES_PER_WAVE;
  _Float16* embs   = (_Float16*)(smem + LDS_EMB_OFF)    + wave * (ROWS * EMB_STRIDE);
  float*    dech   = (float*)(smem + LDS_DEC_OFF)       + wave * (ROWS * DEC_STRIDE);

  const int r0    = (blockIdx.x * WAVES_PER_BLOCK + wave) * ROWS;
  const int m     = lane & 15;              // A row / D column index
  const int nn    = lane & 15;
  const int mBase = (lane >> 4) * 8;        // D row base
  const int kbase = (lane >> 4) * 8;        // A K-base
  const int cbase = (lane >> 4) * 32;

  // ---- kick off DMA of stage 0 weights immediately ----
  if (wave == 0) tdm_load_stage(ws, 0u);

  // ---- node embedding + node_h staged (f16) into concat cols [0..63], [448..511] ----
  {
    float p0 = node_pos[(size_t)(r0 + m) * 2 + 0];
    float p1 = node_pos[(size_t)(r0 + m) * 2 + 1];
#pragma unroll 8
    for (int c = cbase; c < cbase + 32; ++c)
      concat[m * CONCAT_STRIDE + c] =
          (_Float16)ftanh_(p0 * n_emb_w[c] + p1 * n_emb_w[64 + c] + n_emb_b[c]);
    const float* hrow = node_h + (size_t)(r0 + m) * 64;
#pragma unroll 8
    for (int c = cbase; c < cbase + 32; ++c)
      concat[m * CONCAT_STRIDE + 448 + c] = (_Float16)hrow[c];
  }

  // ==================== edge LSTMs (stages 0..5) ====================
#pragma unroll 1
  for (int e = 0; e < NN_E; ++e) {
    if (wave == 0) __builtin_amdgcn_s_wait_tensorcnt(0);   // stage e DMA complete
    __syncthreads();                                       // all waves: buf ready, prev buf free
    if (wave == 0 && e + 1 < N_STAGES)                     // prefetch stage e+1 into other buf
      tdm_load_stage(ws + (size_t)(e + 1) * STAGE_HALVES,
                     (unsigned)(((e + 1) & 1) * LDS_W_BYTES));
    const _Float16* ldsW = (const _Float16*)(smem + (e & 1) * LDS_W_BYTES);

    // embedding: tanh(x @ Wemb + b) -> f16 LDS [16][64]
    {
      float x0 = edge_inputs[((size_t)e * N + r0 + m) * 2 + 0];
      float x1 = edge_inputs[((size_t)e * N + r0 + m) * 2 + 1];
      const float* w0 = e_emb_w + (size_t)e * 128;
      const float* w1 = w0 + 64;
      const float* bb = e_emb_b + (size_t)e * 64;
#pragma unroll 8
      for (int c = cbase; c < cbase + 32; ++c)
        embs[m * EMB_STRIDE + c] = (_Float16)ftanh_(x0 * w0[c] + x1 * w1[c] + bb[c]);
    }

    // A fragments (K=64 -> 2 steps) for emb (LDS) and h (global f32 -> f16)
    const float* hrow = edge_h + ((size_t)e * N + r0 + m) * 64;
    v16h hA0 = load_afrag_gf32(hrow, 0, kbase);
    v16h hA1 = load_afrag_gf32(hrow, 1, kbase);
    v16h eA0 = load_afrag_lds(embs + m * EMB_STRIDE + 0 * 32 + kbase);
    v16h eA1 = load_afrag_lds(embs + m * EMB_STRIDE + 1 * 32 + kbase);

#pragma unroll 1
    for (int hct = 0; hct < 4; ++hct) {
      v8f acc[4];
#pragma unroll
      for (int g = 0; g < 4; ++g) {
        int ct = g * 4 + hct;
        float bias = e_bih[(size_t)e * 256 + ct * 16 + nn] +
                     e_bhh[(size_t)e * 256 + ct * 16 + nn];
        v8f a = splat8(bias);
        a = WMMA(eA0, load_bfrag(ldsW, (ct * 2 + 0) * 2 + 0, lane), a);
        a = WMMA(eA1, load_bfrag(ldsW, (ct * 2 + 0) * 2 + 1, lane), a);
        a = WMMA(hA0, load_bfrag(ldsW, (ct * 2 + 1) * 2 + 0, lane), a);
        a = WMMA(hA1, load_bfrag(ldsW, (ct * 2 + 1) * 2 + 1, lane), a);
        acc[g] = a;
      }
      // cell update for hidden slice [hct*16 .. hct*16+15]
#pragma unroll
      for (int r = 0; r < 8; ++r) {
        float iv = fsig(acc[0][r]);
        float fv = fsig(acc[1][r]);
        float gv = ftanh_(acc[2][r]);
        float ov = fsig(acc[3][r]);
        size_t idx = ((size_t)e * N + r0 + mBase + r) * 64 + hct * 16 + nn;
        float c1 = fv * edge_c[idx] + iv * gv;
        float h1 = ov * ftanh_(c1);
        out_ec[idx] = c1;
        out_eh[idx] = h1;
        concat[(mBase + r) * CONCAT_STRIDE + 64 + e * 64 + hct * 16 + nn] = (_Float16)h1;
      }
    }
  }

  // ==================== node LSTM, fused K=512 (stages 6..9) ====================
#pragma unroll 1
  for (int hct = 0; hct < 4; ++hct) {
    const int s = NN_E + hct;
    if (wave == 0) __builtin_amdgcn_s_wait_tensorcnt(0);
    __syncthreads();
    if (wave == 0 && s + 1 < N_STAGES)
      tdm_load_stage(ws + (size_t)(s + 1) * STAGE_HALVES,
                     (unsigned)(((s + 1) & 1) * LDS_W_BYTES));
    const _Float16* ldsW = (const _Float16*)(smem + (s & 1) * LDS_W_BYTES);

    v8f acc[4];
#pragma unroll
    for (int g = 0; g < 4; ++g) {
      int ct = g * 4 + hct;
      acc[g] = splat8(n_bih[ct * 16 + nn] + n_bhh[ct * 16 + nn]);
    }
#pragma unroll
    for (int kk = 0; kk < 16; ++kk) {
      v16h aF = load_afrag_lds(concat + m * CONCAT_STRIDE + kk * 32 + kbase);
#pragma unroll
      for (int g = 0; g < 4; ++g)
        acc[g] = WMMA(aF, load_bfrag(ldsW, g * 16 + kk, lane), acc[g]);
    }
#pragma unroll
    for (int r = 0; r < 8; ++r) {
      float iv = fsig(acc[0][r]);
      float fv = fsig(acc[1][r]);
      float gv = ftanh_(acc[2][r]);
      float ov = fsig(acc[3][r]);
      size_t idx = (size_t)(r0 + mBase + r) * 64 + hct * 16 + nn;
      float c1 = fv * node_c[idx] + iv * gv;
      float h1 = ov * ftanh_(c1);
      out_nc[idx] = c1;
      out_nh[idx] = h1;
      dech[(mBase + r) * DEC_STRIDE + hct * 16 + nn] = h1;
    }
  }

  // ==================== decoder + residual ====================
  if (lane < 16) {
    int mm = lane;
    float p0 = node_pos[(size_t)(r0 + mm) * 2 + 0];
    float p1 = node_pos[(size_t)(r0 + mm) * 2 + 1];
    float d0 = dec_b[0], d1 = dec_b[1];
#pragma unroll 16
    for (int h = 0; h < 64; ++h) {
      float v = dech[mm * DEC_STRIDE + h];
      d0 += v * dec_w[h * 2 + 0];
      d1 += v * dec_w[h * 2 + 1];
    }
    out_pos[(size_t)(r0 + mm) * 2 + 0] = p0 + ftanh_(d0);
    out_pos[(size_t)(r0 + mm) * 2 + 1] = p1 + ftanh_(d1);
  }
}

extern "C" void kernel_launch(void* const* d_in, const int* in_sizes, int n_in,
                              void* d_out, int out_size, void* d_ws, size_t ws_size,
                              hipStream_t stream) {
  const float* node_pos    = (const float*)d_in[0];
  const float* edge_inputs = (const float*)d_in[1];
  const float* edge_h      = (const float*)d_in[2];
  const float* edge_c      = (const float*)d_in[3];
  const float* node_h      = (const float*)d_in[4];
  const float* node_c      = (const float*)d_in[5];
  const float* e_emb_w     = (const float*)d_in[6];
  const float* e_emb_b     = (const float*)d_in[7];
  const float* e_wih       = (const float*)d_in[8];
  const float* e_whh       = (const float*)d_in[9];
  const float* e_bih       = (const float*)d_in[10];
  const float* e_bhh       = (const float*)d_in[11];
  const float* n_emb_w     = (const float*)d_in[12];
  const float* n_emb_b     = (const float*)d_in[13];
  const float* n_wih       = (const float*)d_in[14];
  const float* n_whh       = (const float*)d_in[15];
  const float* n_bih       = (const float*)d_in[16];
  const float* n_bhh       = (const float*)d_in[17];
  const float* dec_w       = (const float*)d_in[18];
  const float* dec_b       = (const float*)d_in[19];

  const int N = in_sizes[0] / 2;            // 131072
  _Float16* ws = (_Float16*)d_ws;

  float* out_pos = (float*)d_out;
  float* out_eh  = out_pos + (size_t)2 * N;
  float* out_ec  = out_eh + (size_t)NN_E * N * 64;
  float* out_nh  = out_ec + (size_t)NN_E * N * 64;
  float* out_nc  = out_nh + (size_t)N * 64;

  mesrnn_pack_weights<<<(WS_TOTAL_HALVES + 255) / 256, 256, 0, stream>>>(
      e_wih, e_whh, n_wih, n_whh, ws);

  const int blocks = N / (WAVES_PER_BLOCK * ROWS);   // 1024
  mesrnn_main<<<blocks, BLOCK_THREADS, LDS_TOTAL, stream>>>(
      node_pos, edge_inputs, edge_h, edge_c, node_h, node_c,
      e_emb_w, e_emb_b, e_bih, e_bhh, n_emb_w, n_emb_b, n_bih, n_bhh,
      dec_w, dec_b, ws, out_pos, out_eh, out_ec, out_nh, out_nc, N);
}
